// PerformerMultiHeadAttention_13280038879472
// MI455X (gfx1250) — matrix operand order repaired and checked by reading, not compiled, not run on hardware
//
#include <hip/hip_runtime.h>
#include <hip/hip_bf16.h>
#include <math.h>

// Performer (FAVOR+) attention, fused for MI455X (gfx1250).
// B=4, H=16, N=4096, d=64, m=266 (padded to 288 = 9*32 WMMA K-steps).
// GEMMs on v_wmma_f32_16x16x32_bf16 (bf16 in, fp32 accum), with two
// independent accumulator chains interleaved per wave to cover the
// WMMA->WMMA RAW hazard (ISA 7.12.1: 1 NOP + 4 coexec for bf16).
// kp/qp feature maps never touch HBM; context accumulator lives in LDS.
// X tiles staged into LDS by the Tensor Data Mover (double-buffered).

#define FEATD  64
#define NFEAT  266
#define JPAD   288
#define SEQN   4096
#define NTILES 64
#define BHCNT  64
#define CNORM  0.35355339059327373f   // 64^-0.25

typedef __attribute__((ext_vector_type(16))) __bf16 v16bf;
typedef __attribute__((ext_vector_type(8)))  float  v8f;

__device__ __forceinline__ v8f wmma_bf16(v16bf a, v16bf b, v8f c) {
  return __builtin_amdgcn_wmma_f32_16x16x32_bf16(false, a, false, b, (short)0, c,
                                                 false, false);
}

// ---- fragment loaders (ISA 7.12.2, wave32) -------------------------------
__device__ __forceinline__ v16bf frag_a(const __bf16* A, int ld, int lane) {
  const __bf16* row = A + (lane & 15) * ld + ((lane >> 4) << 4);
  v16bf r;
#pragma unroll
  for (int i = 0; i < 16; ++i) r[i] = row[i];
  return r;
}
__device__ __forceinline__ v16bf frag_bT(const __bf16* Bt, int ld, int lane) {
  const __bf16* row = Bt + (lane & 15) * ld + ((lane >> 4) << 4);
  v16bf r;
#pragma unroll
  for (int i = 0; i < 16; ++i) r[i] = row[i];
  return r;
}
__device__ __forceinline__ v16bf frag_b(const __bf16* B, int ld, int lane) {
  const __bf16* col = B + (lane & 15) + (((lane >> 4) << 4) * ld);
  v16bf r;
#pragma unroll
  for (int i = 0; i < 16; ++i) r[i] = col[i * ld];
  return r;
}

// Dual-chain 16x16 tile GEMM over K=64: A row-major [64][64], B = Bt^T with
// Bt row-major [rows][64]. Two independent accumulator chains interleave so
// the second chain's WMMA + fragment loads fill the first chain's RAW slots.
template <bool DUAL>
__device__ __forceinline__ void gemm2_64(const __bf16* Abase, int mt0, int mt1,
                                         const __bf16* Bbase, int jt0, int jt1,
                                         int lane, v8f& acc0, v8f& acc1) {
  const __bf16* A0 = Abase + mt0 * 16 * 64;
  const __bf16* B0 = Bbase + jt0 * 16 * 64;
  const __bf16* A1 = Abase + mt1 * 16 * 64;
  const __bf16* B1 = Bbase + jt1 * 16 * 64;
  acc0 = wmma_bf16(frag_a(A0, 64, lane), frag_bT(B0, 64, lane), acc0);
  if (DUAL) acc1 = wmma_bf16(frag_a(A1, 64, lane), frag_bT(B1, 64, lane), acc1);
  acc0 = wmma_bf16(frag_a(A0 + 32, 64, lane), frag_bT(B0 + 32, 64, lane), acc0);
  if (DUAL) acc1 = wmma_bf16(frag_a(A1 + 32, 64, lane), frag_bT(B1 + 32, 64, lane), acc1);
}

// order-independent (deterministic) float max via CAS
__device__ __forceinline__ void atomic_max_f32(float* addr, float val) {
  unsigned* ua = (unsigned*)addr;
  unsigned cur = __float_as_uint(*addr);
  while (__uint_as_float(cur) < val) {
    unsigned assumed = cur;
    cur = atomicCAS(ua, assumed, __float_as_uint(val));
    if (cur == assumed) break;
  }
}

// ---- Tensor Data Mover staging (gfx1250) ---------------------------------
#if defined(__gfx1250__) && __has_builtin(__builtin_amdgcn_tensor_load_to_lds)
#define HAVE_TDM 1
#else
#define HAVE_TDM 0
#endif

#if HAVE_TDM
typedef unsigned int u32;
typedef unsigned int u32x4 __attribute__((ext_vector_type(4)));
typedef int i32x4 __attribute__((ext_vector_type(4)));
typedef int i32x8 __attribute__((ext_vector_type(8)));
#define AS3 __attribute__((address_space(3)))

__device__ __forceinline__ u32 lds_off(const void* p) {
  return (u32)(__SIZE_TYPE__)(AS3 const char*)p;   // addrspacecast -> LDS offset
}

// DMA one contiguous 64x64 fp32 tile (16KB) into LDS. D# per ISA ch.8:
// group0: count=1 | lds_addr | global_addr | type=2
// group1: data_size=4B, tensor_dim0/1=64, tile_dim0/1=64, dim0_stride=64
__device__ __forceinline__ void tdm_load_tile(const float* gsrc, u32 ldsaddr) {
  unsigned long long ga = (unsigned long long)(__SIZE_TYPE__)gsrc;
  u32x4 g0 = { 1u, ldsaddr, (u32)ga, (u32)(ga >> 32) | 0x80000000u };
  i32x8 g1 = { 0x20000, (int)(64u << 16), (int)(64u << 16), (int)(64u << 16),
               64, 64, 0, 0 };
  i32x4 gz = { 0, 0, 0, 0 };
#if defined(__clang_major__) && (__clang_major__ >= 23)
  i32x8 gz8 = { 0, 0, 0, 0, 0, 0, 0, 0 };
  __builtin_amdgcn_tensor_load_to_lds(g0, g1, gz, gz, gz8, 0);
#else
  __builtin_amdgcn_tensor_load_to_lds(g0, g1, gz, gz, 0);
#endif
}

__device__ __forceinline__ void wait_tensorcnt0() {
#if __has_builtin(__builtin_amdgcn_s_wait_tensorcnt)
  __builtin_amdgcn_s_wait_tensorcnt(0);
#else
  asm volatile("s_wait_tensorcnt 0x0" ::: "memory");
#endif
}
#endif  // HAVE_TDM

// ---- kernel 0: init key stabilizers --------------------------------------
__global__ void k_init(float* __restrict__ stabk) {
  if (threadIdx.x < BHCNT) stabk[threadIdx.x] = -__builtin_inff();
}

// ---- kernel 1: per-head max of key feature logits ------------------------
__global__ __launch_bounds__(256) void k_stabk(const float* __restrict__ x,
                                               const float* __restrict__ Wkw,
                                               const float* __restrict__ proj,
                                               float* __restrict__ stabk) {
  extern __shared__ char smem[];
  __bf16* sW  = (__bf16*)smem;        // 64x64
  __bf16* sP  = sW + 64 * 64;         // JPAD x 64 (rows >= 266 zero)
  __bf16* sX  = sP + JPAD * 64;       // 64x64
  __bf16* sKc = sX + 64 * 64;         // 64x64: c * (X Wk^T)
  float*  red = (float*)(sKc + 64 * 64);

  const int t = threadIdx.x, lane = t & 31, wave = t >> 5;
  const int tile = blockIdx.x & 63, bh = blockIdx.x >> 6;

  { const float4* w4 = (const float4*)Wkw;
    for (int i = t; i < 64 * 16; i += 256) {
      float4 w = w4[i]; int o = i * 4;
      sW[o] = (__bf16)w.x; sW[o+1] = (__bf16)w.y; sW[o+2] = (__bf16)w.z; sW[o+3] = (__bf16)w.w;
    } }
  { const float4* p4 = (const float4*)proj;
    for (int i = t; i < JPAD * 16; i += 256) {
      float4 w{0.f, 0.f, 0.f, 0.f};
      if ((i >> 4) < NFEAT) w = p4[i];
      int o = i * 4;
      sP[o] = (__bf16)w.x; sP[o+1] = (__bf16)w.y; sP[o+2] = (__bf16)w.z; sP[o+3] = (__bf16)w.w;
    } }
  { const float4* x4 = (const float4*)(x + (((size_t)bh * SEQN) + (size_t)tile * 64) * 64);
    for (int i = t; i < 64 * 16; i += 256) {
      float4 w = x4[i]; int o = i * 4;
      sX[o] = (__bf16)w.x; sX[o+1] = (__bf16)w.y; sX[o+2] = (__bf16)w.z; sX[o+3] = (__bf16)w.w;
    } }
  __syncthreads();

  // Kc = c * X * Wk^T  (2 jobs per wave, dual-chain)
  {
    int job0 = wave, job1 = wave + 8;
    v8f acc0 = {}, acc1 = {};
    gemm2_64<true>(sX, job0 >> 2, job1 >> 2, sW, job0 & 3, job1 & 3, lane, acc0, acc1);
    int col0 = (job0 & 3) * 16 + (lane & 15), rb0 = (job0 >> 2) * 16 + ((lane >> 4) << 3);
    int col1 = (job1 & 3) * 16 + (lane & 15), rb1 = (job1 >> 2) * 16 + ((lane >> 4) << 3);
#pragma unroll
    for (int g = 0; g < 8; ++g) {
      sKc[(rb0 + g) * 64 + col0] = (__bf16)(acc0[g] * CNORM);
      sKc[(rb1 + g) * 64 + col1] = (__bf16)(acc1[g] * CNORM);
    }
  }
  __syncthreads();

  // Dk = Kc * P^T ; max over valid features (4 dual pairs + 1 single)
  float mx = -__builtin_inff();
#pragma unroll
  for (int pp = 0; pp < 5; ++pp) {
    int job0 = wave + 16 * pp, job1 = job0 + 8;
    int mt0 = job0 / 18, jt0 = job0 % 18;
    int mt1 = job1 / 18, jt1 = job1 % 18;
    v8f acc0 = {}, acc1 = {};
    if (pp < 4) gemm2_64<true>(sKc, mt0, mt1, sP, jt0, jt1, lane, acc0, acc1);
    else        gemm2_64<false>(sKc, mt0, mt1, sP, jt0, jt1, lane, acc0, acc1);
    int j0 = jt0 * 16 + (lane & 15);
    if (j0 < NFEAT) {
#pragma unroll
      for (int g = 0; g < 8; ++g) mx = fmaxf(mx, acc0[g]);
    }
    if (pp < 4) {
      int j1 = jt1 * 16 + (lane & 15);
      if (j1 < NFEAT) {
#pragma unroll
        for (int g = 0; g < 8; ++g) mx = fmaxf(mx, acc1[g]);
      }
    }
  }
#pragma unroll
  for (int off = 16; off > 0; off >>= 1) mx = fmaxf(mx, __shfl_xor(mx, off));
  if (lane == 0) red[wave] = mx;
  __syncthreads();
  if (t == 0) {
    float m = red[0];
    for (int w = 1; w < 8; ++w) m = fmaxf(m, red[w]);
    atomic_max_f32(&stabk[bh], m);
  }
}

// ---- kernel 2: kp, ksum, context (one workgroup owns one head) -----------
__global__ __launch_bounds__(256) void k_ctx(const float* __restrict__ x,
                                             const float* __restrict__ Wkw,
                                             const float* __restrict__ Wvw,
                                             const float* __restrict__ proj,
                                             const float* __restrict__ stabk,
                                             float* __restrict__ ksum_g,
                                             float* __restrict__ ctx_g) {
  extern __shared__ char smem[];
  __bf16* sWk = (__bf16*)smem;          // 64x64
  __bf16* sWv = sWk + 64 * 64;          // 64x64
  __bf16* sP  = sWv + 64 * 64;          // JPADx64
  __bf16* sX  = sP + JPAD * 64;         // 64x64 bf16
  __bf16* sKc = sX + 64 * 64;           // 64x64
  __bf16* sV  = sKc + 64 * 64;          // 64x64
  __bf16* sKpT = sV + 64 * 64;          // TRANSPOSED kp: [JPAD][64]
  float* sCtx  = (float*)(sKpT + JPAD * 64); // JPAD x 64 fp32 accumulator
  float* sKsum = sCtx + JPAD * 64;           // JPAD
  float* sDiag = sKsum + JPAD;               // 64
  float* sXf   = sDiag + 64;                 // 2 x 4096 fp32 TDM staging

  const int t = threadIdx.x, lane = t & 31, wave = t >> 5;
  const int bh = blockIdx.x;
  const float ratio = rsqrtf(266.0f);
  const float stab = stabk[bh];

  { const float4* a4 = (const float4*)Wkw; const float4* b4 = (const float4*)Wvw;
    for (int i = t; i < 64 * 16; i += 256) {
      float4 wa = a4[i], wb = b4[i]; int o = i * 4;
      sWk[o] = (__bf16)wa.x; sWk[o+1] = (__bf16)wa.y; sWk[o+2] = (__bf16)wa.z; sWk[o+3] = (__bf16)wa.w;
      sWv[o] = (__bf16)wb.x; sWv[o+1] = (__bf16)wb.y; sWv[o+2] = (__bf16)wb.z; sWv[o+3] = (__bf16)wb.w;
    } }
  { const float4* p4 = (const float4*)proj;
    for (int i = t; i < JPAD * 16; i += 256) {
      float4 w{0.f, 0.f, 0.f, 0.f};
      if ((i >> 4) < NFEAT) w = p4[i];
      int o = i * 4;
      sP[o] = (__bf16)w.x; sP[o+1] = (__bf16)w.y; sP[o+2] = (__bf16)w.z; sP[o+3] = (__bf16)w.w;
    } }
  for (int i = t; i < JPAD * 64; i += 256) sCtx[i] = 0.f;
  for (int i = t; i < JPAD; i += 256) sKsum[i] = 0.f;

  const float* xbase = x + ((size_t)bh * SEQN) * 64;
#if HAVE_TDM
  const u32 sxf_base = lds_off(sXf);
  if (wave == 0) tdm_load_tile(xbase, sxf_base);   // tile 0 -> buffer 0
#endif
  int buf = 0;

  for (int tile = 0; tile < NTILES; ++tile) {
#if HAVE_TDM
    if (wave == 0) wait_tensorcnt0();
    __syncthreads();                       // DMA done + prev compute done
    { const float4* s4 = (const float4*)(sXf + buf * 4096);
      for (int i = t; i < 64 * 16; i += 256) {
        float4 w = s4[i]; int o = i * 4;
        sX[o] = (__bf16)w.x; sX[o+1] = (__bf16)w.y; sX[o+2] = (__bf16)w.z; sX[o+3] = (__bf16)w.w;
      } }
    if (wave == 0 && tile + 1 < NTILES)    // DMA next tile into other buffer
      tdm_load_tile(xbase + (size_t)(tile + 1) * 4096,
                    sxf_base + (u32)((buf ^ 1) * 16384));
    __syncthreads();
#else
    __syncthreads();
    { const float4* x4 = (const float4*)(xbase + (size_t)tile * 4096);
      if (tile + 1 < NTILES) __builtin_prefetch(xbase + (size_t)(tile + 1) * 4096, 0, 1);
      for (int i = t; i < 64 * 16; i += 256) {
        float4 w = x4[i]; int o = i * 4;
        sX[o] = (__bf16)w.x; sX[o+1] = (__bf16)w.y; sX[o+2] = (__bf16)w.z; sX[o+3] = (__bf16)w.w;
      } }
    __syncthreads();
#endif

    // K (scaled) then V tiles: 2 dual pairs per wave
#pragma unroll
    for (int pp = 0; pp < 2; ++pp) {
      int job0 = wave + 16 * pp, job1 = job0 + 8;       // both K (pp=0) or V (pp=1)
      const __bf16* W = pp ? sWv : sWk;
      int mt0 = (job0 >> 2) & 3, nt0 = job0 & 3;
      int mt1 = (job1 >> 2) & 3, nt1 = job1 & 3;
      v8f acc0 = {}, acc1 = {};
      gemm2_64<true>(sX, mt0, mt1, W, nt0, nt1, lane, acc0, acc1);
      int col0 = nt0 * 16 + (lane & 15), rb0 = mt0 * 16 + ((lane >> 4) << 3);
      int col1 = nt1 * 16 + (lane & 15), rb1 = mt1 * 16 + ((lane >> 4) << 3);
      if (pp == 0) {
#pragma unroll
        for (int g = 0; g < 8; ++g) {
          sKc[(rb0 + g) * 64 + col0] = (__bf16)(acc0[g] * CNORM);
          sKc[(rb1 + g) * 64 + col1] = (__bf16)(acc1[g] * CNORM);
        }
      } else {
#pragma unroll
        for (int g = 0; g < 8; ++g) {
          sV[(rb0 + g) * 64 + col0] = (__bf16)acc0[g];
          sV[(rb1 + g) * 64 + col1] = (__bf16)acc1[g];
        }
      }
    }
    __syncthreads();
    if (t < 64) {  // diag = 0.5 * ||c*K_n||^2
      float s = 0.f;
      for (int d = 0; d < 64; ++d) { float kv = (float)sKc[t * 64 + d]; s += kv * kv; }
      sDiag[t] = 0.5f * s;
    }
    __syncthreads();

    // kp = ratio*(exp(Dk - diag - stab) + eps); stored TRANSPOSED [j][n]
#pragma unroll
    for (int pp = 0; pp < 5; ++pp) {
      int job0 = wave + 16 * pp, job1 = job0 + 8;
      int mt0 = job0 / 18, jt0 = job0 % 18;
      int mt1 = job1 / 18, jt1 = job1 % 18;
      v8f acc0 = {}, acc1 = {};
      if (pp < 4) gemm2_64<true>(sKc, mt0, mt1, sP, jt0, jt1, lane, acc0, acc1);
      else        gemm2_64<false>(sKc, mt0, mt1, sP, jt0, jt1, lane, acc0, acc1);
      {
        int j = jt0 * 16 + (lane & 15), rb = mt0 * 16 + ((lane >> 4) << 3);
        bool valid = (j < NFEAT);
#pragma unroll
        for (int g = 0; g < 8; ++g) {
          float v = valid ? (ratio * (__expf(acc0[g] - sDiag[rb + g] - stab) + 1e-4f)) : 0.f;
          sKpT[j * 64 + rb + g] = (__bf16)v;
        }
      }
      if (pp < 4) {
        int j = jt1 * 16 + (lane & 15), rb = mt1 * 16 + ((lane >> 4) << 3);
        bool valid = (j < NFEAT);
#pragma unroll
        for (int g = 0; g < 8; ++g) {
          float v = valid ? (ratio * (__expf(acc1[g] - sDiag[rb + g] - stab) + 1e-4f)) : 0.f;
          sKpT[j * 64 + rb + g] = (__bf16)v;
        }
      }
    }
    __syncthreads();

    // ksum: deterministic serial sums (unit-stride in transposed layout)
    for (int j = t; j < JPAD; j += 256) {
      float s = 0.f;
      for (int r = 0; r < 64; ++r) s += (float)sKpT[j * 64 + r];
      sKsum[j] += s;
    }

    // ctx += kp^T * V (A = sKpT row-major [j][n]); dual-chain pairs
#pragma unroll
    for (int pp = 0; pp < 5; ++pp) {
      int job0 = wave + 16 * pp, job1 = job0 + 8;
      int jt0 = job0 / 4, dt0 = job0 & 3;
      int jt1 = job1 / 4, dt1 = job1 & 3;
      int col0 = dt0 * 16 + (lane & 15), rb0 = jt0 * 16 + ((lane >> 4) << 3);
      int col1 = dt1 * 16 + (lane & 15), rb1 = jt1 * 16 + ((lane >> 4) << 3);
      v8f acc0 = {}, acc1 = {};
#pragma unroll
      for (int g = 0; g < 8; ++g) acc0[g] = sCtx[(rb0 + g) * 64 + col0];
      if (pp < 4) {
#pragma unroll
        for (int g = 0; g < 8; ++g) acc1[g] = sCtx[(rb1 + g) * 64 + col1];
      }
      const __bf16* A0 = sKpT + jt0 * 16 * 64;
      const __bf16* A1 = sKpT + jt1 * 16 * 64;
      const __bf16* B0 = sV + dt0 * 16;
      const __bf16* B1 = sV + dt1 * 16;
      acc0 = wmma_bf16(frag_a(A0, 64, lane), frag_b(B0, 64, lane), acc0);
      if (pp < 4) acc1 = wmma_bf16(frag_a(A1, 64, lane), frag_b(B1, 64, lane), acc1);
      acc0 = wmma_bf16(frag_a(A0 + 32, 64, lane), frag_b(B0 + 32 * 64, 64, lane), acc0);
      if (pp < 4) acc1 = wmma_bf16(frag_a(A1 + 32, 64, lane), frag_b(B1 + 32 * 64, 64, lane), acc1);
#pragma unroll
      for (int g = 0; g < 8; ++g) sCtx[(rb0 + g) * 64 + col0] = acc0[g];
      if (pp < 4) {
#pragma unroll
        for (int g = 0; g < 8; ++g) sCtx[(rb1 + g) * 64 + col1] = acc1[g];
      }
    }
    buf ^= 1;
  }
  __syncthreads();
  for (int i = t; i < JPAD; i += 256) ksum_g[bh * JPAD + i] = sKsum[i];
  float* cg = ctx_g + (size_t)bh * JPAD * 64;
  for (int i = t; i < JPAD * 64; i += 256) cg[i] = sCtx[i];
}

// ---- kernel 3: qp and output ---------------------------------------------
__global__ __launch_bounds__(256) void k_out(const float* __restrict__ x,
                                             const float* __restrict__ Wqw,
                                             const float* __restrict__ proj,
                                             const float* __restrict__ ksum_g,
                                             const float* __restrict__ ctx_g,
                                             float* __restrict__ out) {
  extern __shared__ char smem[];
  __bf16* sW   = (__bf16*)smem;         // 64x64
  __bf16* sPC  = sW + 64 * 64;          // JPADx64: proj, later reused for ctx
  __bf16* sX   = sPC + JPAD * 64;       // 64x64
  __bf16* sQc  = sX + 64 * 64;          // 64x64
  __bf16* sQp  = sQc + 64 * 64;         // 64 x JPAD (row-major for out GEMM)
  float* sKsum = (float*)(sQp + 64 * JPAD); // JPAD
  float* sDiag = sKsum + JPAD;              // 64
  float* sRMax = sDiag + 64;                // 64
  float* sDen  = sRMax + 64;                // 64

  const int t = threadIdx.x, lane = t & 31, wave = t >> 5;
  const int tile = blockIdx.x & 63, bh = blockIdx.x >> 6;
  const float ratio = rsqrtf(266.0f);

  { const float4* w4 = (const float4*)Wqw;
    for (int i = t; i < 64 * 16; i += 256) {
      float4 w = w4[i]; int o = i * 4;
      sW[o] = (__bf16)w.x; sW[o+1] = (__bf16)w.y; sW[o+2] = (__bf16)w.z; sW[o+3] = (__bf16)w.w;
    } }
  { const float4* p4 = (const float4*)proj;
    for (int i = t; i < JPAD * 16; i += 256) {
      float4 w{0.f, 0.f, 0.f, 0.f};
      if ((i >> 4) < NFEAT) w = p4[i];
      int o = i * 4;
      sPC[o] = (__bf16)w.x; sPC[o+1] = (__bf16)w.y; sPC[o+2] = (__bf16)w.z; sPC[o+3] = (__bf16)w.w;
    } }
  { const float4* x4 = (const float4*)(x + (((size_t)bh * SEQN) + (size_t)tile * 64) * 64);
    for (int i = t; i < 64 * 16; i += 256) {
      float4 w = x4[i]; int o = i * 4;
      sX[o] = (__bf16)w.x; sX[o+1] = (__bf16)w.y; sX[o+2] = (__bf16)w.z; sX[o+3] = (__bf16)w.w;
    } }
  for (int i = t; i < JPAD; i += 256) sKsum[i] = ksum_g[bh * JPAD + i];
  if (t < 64) sRMax[t] = -__builtin_inff();
  __syncthreads();

  // Qc = c * X * Wq^T  (dual-chain)
  {
    int job0 = wave, job1 = wave + 8;
    v8f acc0 = {}, acc1 = {};
    gemm2_64<true>(sX, job0 >> 2, job1 >> 2, sW, job0 & 3, job1 & 3, lane, acc0, acc1);
    int col0 = (job0 & 3) * 16 + (lane & 15), rb0 = (job0 >> 2) * 16 + ((lane >> 4) << 3);
    int col1 = (job1 & 3) * 16 + (lane & 15), rb1 = (job1 >> 2) * 16 + ((lane >> 4) << 3);
#pragma unroll
    for (int g = 0; g < 8; ++g) {
      sQc[(rb0 + g) * 64 + col0] = (__bf16)(acc0[g] * CNORM);
      sQc[(rb1 + g) * 64 + col1] = (__bf16)(acc1[g] * CNORM);
    }
  }
  __syncthreads();
  if (t < 64) {
    float s = 0.f;
    for (int d = 0; d < 64; ++d) { float qv = (float)sQc[t * 64 + d]; s += qv * qv; }
    sDiag[t] = 0.5f * s;
  }
  __syncthreads();

  // pass 1: per-row max of Dq (order-independent atomic max)
#pragma unroll
  for (int pp = 0; pp < 5; ++pp) {
    int job0 = wave + 16 * pp, job1 = job0 + 8;
    int mt0 = job0 / 18, jt0 = job0 % 18;
    int mt1 = job1 / 18, jt1 = job1 % 18;
    v8f acc0 = {}, acc1 = {};
    if (pp < 4) gemm2_64<true>(sQc, mt0, mt1, sPC, jt0, jt1, lane, acc0, acc1);
    else        gemm2_64<false>(sQc, mt0, mt1, sPC, jt0, jt1, lane, acc0, acc1);
    {
      int j = jt0 * 16 + (lane & 15), rb = mt0 * 16 + ((lane >> 4) << 3);
      if (j < NFEAT) {
#pragma unroll
        for (int g = 0; g < 8; ++g) atomic_max_f32(&sRMax[rb + g], acc0[g]);
      }
    }
    if (pp < 4) {
      int j = jt1 * 16 + (lane & 15), rb = mt1 * 16 + ((lane >> 4) << 3);
      if (j < NFEAT) {
#pragma unroll
        for (int g = 0; g < 8; ++g) atomic_max_f32(&sRMax[rb + g], acc1[g]);
      }
    }
  }
  __syncthreads();

  // pass 2: qp (recompute Dq, bit-identical to pass 1)
#pragma unroll
  for (int pp = 0; pp < 5; ++pp) {
    int job0 = wave + 16 * pp, job1 = job0 + 8;
    int mt0 = job0 / 18, jt0 = job0 % 18;
    int mt1 = job1 / 18, jt1 = job1 % 18;
    v8f acc0 = {}, acc1 = {};
    if (pp < 4) gemm2_64<true>(sQc, mt0, mt1, sPC, jt0, jt1, lane, acc0, acc1);
    else        gemm2_64<false>(sQc, mt0, mt1, sPC, jt0, jt1, lane, acc0, acc1);
    {
      int j = jt0 * 16 + (lane & 15), rb = mt0 * 16 + ((lane >> 4) << 3);
      bool valid = (j < NFEAT);
#pragma unroll
      for (int g = 0; g < 8; ++g) {
        float v = valid
          ? (ratio * (__expf(acc0[g] - sDiag[rb + g] - sRMax[rb + g]) + 1e-4f)) : 0.f;
        sQp[(rb + g) * JPAD + j] = (__bf16)v;
      }
    }
    if (pp < 4) {
      int j = jt1 * 16 + (lane & 15), rb = mt1 * 16 + ((lane >> 4) << 3);
      bool valid = (j < NFEAT);
#pragma unroll
      for (int g = 0; g < 8; ++g) {
        float v = valid
          ? (ratio * (__expf(acc1[g] - sDiag[rb + g] - sRMax[rb + g]) + 1e-4f)) : 0.f;
        sQp[(rb + g) * JPAD + j] = (__bf16)v;
      }
    }
  }
  __syncthreads();

  // denominators + stage context (overwrites proj region)
  if (t < 64) {
    float s = 0.f;
    for (int j = 0; j < JPAD; ++j) s += (float)sQp[t * JPAD + j] * sKsum[j];
    sDen[t] = 1.0f / s;
  }
  { const float4* c4 = (const float4*)(ctx_g + (size_t)bh * JPAD * 64);
    for (int i = t; i < JPAD * 16; i += 256) {
      float4 w = c4[i]; int o = i * 4;
      sPC[o] = (__bf16)w.x; sPC[o+1] = (__bf16)w.y; sPC[o+2] = (__bf16)w.z; sPC[o+3] = (__bf16)w.w;
    } }
  __syncthreads();

  // out = (qp * ctx) * Dinv : two interleaved 9-step K-chains per wave
  float* og = out + (((size_t)bh * SEQN) + (size_t)tile * 64) * 64;
  {
    int job0 = wave, job1 = wave + 8;
    int mt0 = job0 >> 2, dt0 = job0 & 3;
    int mt1 = job1 >> 2, dt1 = job1 & 3;
    v8f acc0 = {}, acc1 = {};
#pragma unroll
    for (int ks = 0; ks < 9; ++ks) {
      acc0 = wmma_bf16(frag_a(sQp + mt0 * 16 * JPAD + ks * 32, JPAD, lane),
                       frag_b(sPC + (ks * 32) * 64 + dt0 * 16, 64, lane), acc0);
      acc1 = wmma_bf16(frag_a(sQp + mt1 * 16 * JPAD + ks * 32, JPAD, lane),
                       frag_b(sPC + (ks * 32) * 64 + dt1 * 16, 64, lane), acc1);
    }
    int col0 = dt0 * 16 + (lane & 15), rb0 = mt0 * 16 + ((lane >> 4) << 3);
    int col1 = dt1 * 16 + (lane & 15), rb1 = mt1 * 16 + ((lane >> 4) << 3);
#pragma unroll
    for (int g = 0; g < 8; ++g) {
      og[(rb0 + g) * 64 + col0] = acc0[g] * sDen[rb0 + g];
      og[(rb1 + g) * 64 + col1] = acc1[g] * sDen[rb1 + g];
    }
  }
}

extern "C" void kernel_launch(void* const* d_in, const int* in_sizes, int n_in,
                              void* d_out, int out_size, void* d_ws, size_t ws_size,
                              hipStream_t stream) {
  (void)in_sizes; (void)n_in; (void)out_size; (void)ws_size;
  const float* x    = (const float*)d_in[0];
  const float* Wq   = (const float*)d_in[1];
  const float* Wk   = (const float*)d_in[2];
  const float* Wv   = (const float*)d_in[3];
  const float* proj = (const float*)d_in[4];
  float* out = (float*)d_out;

  // workspace: [stabk 64][ksum 64*288][ctx 64*288*64] fp32  (~4.8 MB)
  float* stabk = (float*)d_ws;
  float* ksum  = stabk + BHCNT;
  float* ctxg  = ksum + BHCNT * JPAD;

  const size_t smem_stabk = (size_t)(64 * 64 * 3 + JPAD * 64) * 2 + 8 * 4;
  const size_t smem_ctx   = (size_t)(64 * 64 * 5 + JPAD * 64 + JPAD * 64) * 2 +
                            (size_t)(JPAD * 64 + JPAD + 64) * 4 +
                            (size_t)(2 * 4096) * 4;          // TDM staging
  const size_t smem_out   = (size_t)(64 * 64 * 3 + JPAD * 64 + 64 * JPAD) * 2 +
                            (size_t)(JPAD + 64 * 3) * 4;

  k_init<<<1, 64, 0, stream>>>(stabk);
  k_stabk<<<BHCNT * NTILES, 256, smem_stabk, stream>>>(x, Wk, proj, stabk);
  k_ctx<<<BHCNT, 256, smem_ctx, stream>>>(x, Wk, Wv, proj, stabk, ksum, ctxg);
  k_out<<<BHCNT * NTILES, 256, smem_out, stream>>>(x, Wq, proj, ksum, ctxg, out);
}
